// RNNNet_19275813224656
// MI455X (gfx1250) — matrix operand order repaired and checked by reading, not compiled, not run on hardware
//
#include <hip/hip_runtime.h>
#include <hip/hip_bf16.h>

typedef _Float16 v4h  __attribute__((ext_vector_type(4)));
typedef _Float16 v8h  __attribute__((ext_vector_type(8)));
typedef _Float16 v16h __attribute__((ext_vector_type(16)));
typedef float    v8f  __attribute__((ext_vector_type(8)));

#define RNN_IN  28
#define RNN_H   64
#define RNN_OUT 10
#define RNN_T   256
#define RNN_B   4096

#if __has_builtin(__builtin_amdgcn_tanhf)
#define FAST_TANH(x) __builtin_amdgcn_tanhf(x)
#else
#define FAST_TANH(x) tanhf(x)
#endif

// Load a 16x32 f16 A-fragment from a row-major LDS tile.
// Per CDNA5 ISA 16-bit A layout: lanes 0-15 hold row r, K = 0..7 (V0-V3) and
// K = 16..23 (V4-V7); lanes 16-31 hold row r, K = 8..15 and 24..31.
__device__ __forceinline__ v16h lds_load_A(const _Float16* tile, int stride,
                                           int row, int kbase, int half) {
  const _Float16* p = tile + row * stride + kbase + half * 8;
  v8h lo = *(const v8h*)p;
  v8h hi = *(const v8h*)(p + 16);
  return __builtin_shufflevector(lo, hi, 0,1,2,3,4,5,6,7,8,9,10,11,12,13,14,15);
}

// Load a 32x16 f16 B-fragment (B = W^T tile) from row-major f16 weights W[64][stride].
// Dense 16-bit B layout: lane n (0-15) holds column n of B = row (nt*16+n) of W,
// K = 0..7 and 16..23; lanes 16-31 hold K = 8..15 and 24..31 (matches A).
__device__ __forceinline__ v16h lds_load_B(const _Float16* w, int stride,
                                           int nt, int kc, int lane) {
  const int n = lane & 15, half = lane >> 4;
  const _Float16* p = w + (nt * 16 + n) * stride + kc * 32 + half * 8;
  v8h lo = *(const v8h*)p;
  v8h hi = *(const v8h*)(p + 16);
  return __builtin_shufflevector(lo, hi, 0,1,2,3,4,5,6,7,8,9,10,11,12,13,14,15);
}

__device__ __forceinline__ v8f wmma_f16(v16h a, v16h b, v8f c) {
  return __builtin_amdgcn_wmma_f32_16x16x32_f16(false, a, false, b, (short)0, c,
                                                false, false);
}

__global__ __launch_bounds__(64) void rnn2_fused_scan_kernel(
    const float* __restrict__ x,
    const float* __restrict__ Wih0, const float* __restrict__ Whh0,
    const float* __restrict__ bih0, const float* __restrict__ bhh0,
    const float* __restrict__ Wih1, const float* __restrict__ Whh1,
    const float* __restrict__ bih1, const float* __restrict__ bhh1,
    const float* __restrict__ Wfc,  const float* __restrict__ bfc,
    float* __restrict__ out)
{
  // Shared f16 weights (converted once per block)
  __shared__ _Float16 s_wih0[64 * 32];   // K padded 28 -> 32
  __shared__ _Float16 s_whh0[64 * 64];
  __shared__ _Float16 s_wih1[64 * 64];
  __shared__ _Float16 s_whh1[64 * 64];
  // Per-wave staging: x tile (32 rows x 32 K, padded) and ping-pong h tiles
  __shared__ _Float16 s_x [2][32 * 32];
  __shared__ _Float16 s_h0[2][2][32 * 64];
  __shared__ _Float16 s_h1[2][2][32 * 64];

  const int tid  = threadIdx.x;
  const int wave = tid >> 5;
  const int lane = tid & 31;
  const int half = lane >> 4;
  const int r    = lane & 15;
  const int bbase = blockIdx.x * 64 + wave * 32;   // 32 batch rows per wave

  // ---- stage weights f32 -> f16 (cooperative) ----
  for (int i = tid; i < 64 * 32; i += 64) {
    const int row = i >> 5, col = i & 31;
    s_wih0[i] = (_Float16)(col < RNN_IN ? Wih0[row * RNN_IN + col] : 0.0f);
  }
  for (int i = tid; i < 64 * 64; i += 64) {
    s_whh0[i] = (_Float16)Whh0[i];
    s_wih1[i] = (_Float16)Wih1[i];
    s_whh1[i] = (_Float16)Whh1[i];
  }
  // zero own x tile (K pad columns stay 0 forever) and h ping-pong buffer 0 (h_{-1}=0)
  for (int i = lane; i < 32 * 32; i += 32) s_x[wave][i] = (_Float16)0.0f;
  for (int i = lane; i < 32 * 64; i += 32) {
    s_h0[wave][0][i] = (_Float16)0.0f;
    s_h1[wave][0][i] = (_Float16)0.0f;
  }
  __syncthreads();

  // per-lane bias for C-fragment init: element (M, N=lane&15) of N-tile nt
  float bias0v[4], bias1v[4];
#pragma unroll
  for (int nt = 0; nt < 4; ++nt) {
    bias0v[nt] = bih0[nt * 16 + r] + bhh0[nt * 16 + r];
    bias1v[nt] = bih1[nt * 16 + r] + bhh1[nt * 16 + r];
  }

  _Float16* sx = s_x[wave];

  for (int t = 0; t < RNN_T; ++t) {
    const int p = t & 1, q = p ^ 1;

    // ---- stage x tile [32 rows x 28] into LDS f16 (vectorized float4) ----
#pragma unroll
    for (int j = 0; j < 7; ++j) {
      const int i   = lane + 32 * j;       // 0..223 = 32 rows * 7 float4
      const int row = i / 7;
      const int c4  = (i % 7) * 4;
      const float4 v = *(const float4*)(x + ((size_t)(bbase + row) * RNN_T + t) * RNN_IN + c4);
      v4h h4 = { (_Float16)v.x, (_Float16)v.y, (_Float16)v.z, (_Float16)v.w };
      *(v4h*)(sx + row * 32 + c4) = h4;
    }
    // prefetch next timestep's x rows (global_prefetch_b8)
    if (t + 1 < RNN_T) {
      const float* np = x + ((size_t)(bbase + lane) * RNN_T + (t + 1)) * RNN_IN;
      __builtin_prefetch(np, 0, 3);
      __builtin_prefetch((const char*)np + 64, 0, 3);
    }

#pragma unroll
    for (int mt = 0; mt < 2; ++mt) {
      const int arow = mt * 16 + r;

      // ======== layer 0: h0 = tanh(x W_ih0^T + b0 + h0_prev W_hh0^T) ========
      v16h ax  = lds_load_A(sx,             32, arow,  0, half);
      v16h ah0 = lds_load_A(s_h0[wave][p],  64, arow,  0, half);
      v16h ah1 = lds_load_A(s_h0[wave][p],  64, arow, 32, half);
#pragma unroll
      for (int nt = 0; nt < 4; ++nt) {
        v8f acc = (v8f){} + bias0v[nt];
        acc = wmma_f16(ax,  lds_load_B(s_wih0, 32, nt, 0, lane), acc);
        acc = wmma_f16(ah0, lds_load_B(s_whh0, 64, nt, 0, lane), acc);
        acc = wmma_f16(ah1, lds_load_B(s_whh0, 64, nt, 1, lane), acc);
#pragma unroll
        for (int e = 0; e < 8; ++e) {
          const float hv = FAST_TANH(acc[e]);
          s_h0[wave][q][(mt * 16 + e + 8 * half) * 64 + nt * 16 + r] = (_Float16)hv;
        }
      }

      // ======== layer 1: h1 = tanh(h0 W_ih1^T + b1 + h1_prev W_hh1^T) ========
      v16h an0 = lds_load_A(s_h0[wave][q], 64, arow,  0, half);
      v16h an1 = lds_load_A(s_h0[wave][q], 64, arow, 32, half);
      v16h ap0 = lds_load_A(s_h1[wave][p], 64, arow,  0, half);
      v16h ap1 = lds_load_A(s_h1[wave][p], 64, arow, 32, half);
#pragma unroll
      for (int nt = 0; nt < 4; ++nt) {
        v8f acc = (v8f){} + bias1v[nt];
        acc = wmma_f16(an0, lds_load_B(s_wih1, 64, nt, 0, lane), acc);
        acc = wmma_f16(an1, lds_load_B(s_wih1, 64, nt, 1, lane), acc);
        acc = wmma_f16(ap0, lds_load_B(s_whh1, 64, nt, 0, lane), acc);
        acc = wmma_f16(ap1, lds_load_B(s_whh1, 64, nt, 1, lane), acc);
#pragma unroll
        for (int e = 0; e < 8; ++e) {
          const float hv = FAST_TANH(acc[e]);
          s_h1[wave][q][(mt * 16 + e + 8 * half) * 64 + nt * 16 + r] = (_Float16)hv;
        }
      }
    }
  }

  // ---- FC on last timestep: final h1 is in ping-pong buffer ((T-1)&1)^1 == 0 ----
  const _Float16* hf = s_h1[wave][0];
  for (int i = lane; i < 32 * RNN_OUT; i += 32) {
    const int m = i / RNN_OUT, o = i % RNN_OUT;
    float sum = bfc[o];
#pragma unroll
    for (int k = 0; k < RNN_H; ++k)
      sum += (float)hf[m * 64 + k] * Wfc[o * RNN_H + k];
    out[(size_t)(bbase + m) * RNN_OUT + o] = sum;
  }
}

extern "C" void kernel_launch(void* const* d_in, const int* in_sizes, int n_in,
                              void* d_out, int out_size, void* d_ws, size_t ws_size,
                              hipStream_t stream) {
  (void)in_sizes; (void)n_in; (void)out_size; (void)d_ws; (void)ws_size;
  const float* x    = (const float*)d_in[0];
  const float* Wih0 = (const float*)d_in[1];
  const float* Whh0 = (const float*)d_in[2];
  const float* bih0 = (const float*)d_in[3];
  const float* bhh0 = (const float*)d_in[4];
  const float* Wih1 = (const float*)d_in[5];
  const float* Whh1 = (const float*)d_in[6];
  const float* bih1 = (const float*)d_in[7];
  const float* bhh1 = (const float*)d_in[8];
  const float* Wfc  = (const float*)d_in[9];
  const float* bfc  = (const float*)d_in[10];
  float* out = (float*)d_out;

  dim3 grid(RNN_B / 64);   // 64 batch rows per block (2 waves x 32 rows)
  dim3 block(64);
  rnn2_fused_scan_kernel<<<grid, block, 0, stream>>>(
      x, Wih0, Whh0, bih0, bhh0, Wih1, Whh1, bih1, bhh1, Wfc, bfc, out);
}
